// PhiSoftMax_41154376630714
// MI455X (gfx1250) — matrix operand order repaired and checked by reading, not compiled, not run on hardware
//
#include <hip/hip_runtime.h>
#include <hip/hip_bf16.h>

#define B_ 4
#define L_ 1024
#define S_ 1024
#define H_ 8
#define E_ 64
#define PS 1048           // padded f16 row stride (halves) for bank spread
#define EPSV 1e-8f

typedef __attribute__((ext_vector_type(2)))  float     v2f;
typedef __attribute__((ext_vector_type(8)))  float     v8f;
typedef _Float16 half_t;
typedef __attribute__((ext_vector_type(8)))  _Float16  v8h;
typedef __attribute__((ext_vector_type(16))) _Float16  v16h;

// ---- value (B,S,H,E) f32  ->  vt (B,H,E,S) f16 (tiled transpose) ----------
__global__ __launch_bounds__(256) void val_xpose_kernel(
    const float* __restrict__ val, half_t* __restrict__ vt)
{
  __shared__ float tile[32][33];
  const int tx = threadIdx.x & 31;
  const int ty = threadIdx.x >> 5;            // 0..7
  const int blk = blockIdx.x;
  const int et = blk & 1;                     // e tile (2 x 32)
  const int st = (blk >> 1) & 31;             // s tile (32 x 32)
  const int h  = (blk >> 6) & 7;
  const int b  = blk >> 9;
  #pragma unroll
  for (int rr = 0; rr < 4; ++rr) {
    const int s = st * 32 + ty + 8 * rr;
    const int e = et * 32 + tx;
    tile[ty + 8 * rr][tx] = val[(size_t)((b * S_ + s) * H_ + h) * E_ + e];
  }
  __syncthreads();
  #pragma unroll
  for (int rr = 0; rr < 4; ++rr) {
    const int e = et * 32 + ty + 8 * rr;
    const int s = st * 32 + tx;
    vt[(size_t)((b * H_ + h) * E_ + e) * S_ + s] = (half_t)tile[tx][ty + 8 * rr];
  }
}

__global__ __launch_bounds__(256) void phisoftmax_kernel(
    const float* __restrict__ q, const float* __restrict__ key,
    const float* __restrict__ val, const float* __restrict__ pos,
    const float* __restrict__ phi, const float* __restrict__ u,
    const float* __restrict__ p_log_tau, const float* __restrict__ p_thresh,
    const int* __restrict__ p_causal,
    const half_t* __restrict__ vt, const int use_vt,
    float* __restrict__ Vout, float* __restrict__ att, float* __restrict__ ent)
{
  __shared__ __align__(16) _Float16 lds_p[16 * PS];   // normalized probs, f16
  __shared__ float red_max[8][16];
  __shared__ float red_sum[8][16];
  __shared__ float red_ent[8][16];
  __shared__ float lds_red[4][32][8];                 // PV cross-wave partials

  const int tid  = threadIdx.x;
  const int w    = tid >> 5;          // wave id 0..7
  const int lane = tid & 31;
  const int hf   = lane >> 4;         // lane half: 0 or 1
  const int lrow = lane & 15;

  const int blk = blockIdx.x;
  const int qt = blk & 63;            // query tile
  const int h  = blk >> 6;            // head
  const int i0 = qt * 16;

  const float tau     = fminf(fmaxf(__expf(p_log_tau[0]), 0.1f), 5.0f);
  const float inv_tau = __builtin_amdgcn_rcpf(tau);
  const float th      = fminf(fmaxf(p_thresh[0], 0.01f), 0.99f);
  const int  causal   = p_causal[0];
  const float scale   = 0.125f;       // 1/sqrt(64)

  // ---- Phase 0: Gumbel-sigmoid phi-mask, batch-independent, kept in regs ---
  float pm[8][8];
  #pragma unroll
  for (int tt = 0; tt < 8; ++tt) {
    const int j = 16 * (w + 8 * tt) + lrow;
    #pragma unroll
    for (int r = 0; r < 8; ++r) {
      const int i   = i0 + r + 8 * hf;
      const int idx = (h * L_ + i) * S_ + j;
      const float uu = u[idx];
      const float g  = __logf((uu + EPSV) * __builtin_amdgcn_rcpf(1.0f - uu + EPSV));
      const float z  = (g + phi[idx]) * inv_tau;
      const float mr = __builtin_amdgcn_rcpf(1.0f + __expf(-z));
      pm[tt][r] = -10000.0f * fmaxf(0.0f, th - mr);
    }
  }

  // ---------------- batch loop: all phases share pm ----------------
  #pragma unroll 1
  for (int b = 0; b < B_; ++b) {
    // ---- Phase 1: S = Q K^T (fp32 WMMA), logits in regs ----
    v2f afr[16];
    #pragma unroll
    for (int t = 0; t < 16; ++t) {
      const float* qp = q + (size_t)((b * L_ + i0 + lrow) * H_ + h) * E_ + 4 * t + 2 * hf;
      afr[t] = *(const v2f*)qp;
    }
    float p_i[8];
    #pragma unroll
    for (int r = 0; r < 8; ++r) p_i[r] = pos[b * L_ + i0 + r + 8 * hf];

    float logits[8][8];
    #pragma unroll
    for (int tt = 0; tt < 8; ++tt) {
      const int j = 16 * (w + 8 * tt) + lrow;
      const float* kb = key + (size_t)((b * S_ + j) * H_ + h) * E_ + 2 * hf;
      v8f acc = {};
      #pragma unroll
      for (int t = 0; t < 16; ++t) {
        v2f bf = *(const v2f*)(kb + 4 * t);
        acc = __builtin_amdgcn_wmma_f32_16x16x4_f32(false, afr[t], false, bf,
                                                    (short)0, acc, false, false);
      }
      const float p_j = pos[b * L_ + j];
      #pragma unroll
      for (int r = 0; r < 8; ++r) {
        float sc = acc[r] + pm[tt][r];
        if (causal && (p_j > p_i[r])) sc = -__builtin_inff();
        logits[tt][r] = scale * sc;
      }
    }

    // ---- Phase 2: softmax across waves ----
    float rmax[8];
    #pragma unroll
    for (int r = 0; r < 8; ++r) {
      float m = logits[0][r];
      #pragma unroll
      for (int tt = 1; tt < 8; ++tt) m = fmaxf(m, logits[tt][r]);
      #pragma unroll
      for (int off = 1; off < 16; off <<= 1) m = fmaxf(m, __shfl_xor(m, off, 32));
      rmax[r] = m;
    }
    if (lrow == 0) {
      #pragma unroll
      for (int r = 0; r < 8; ++r) red_max[w][r + 8 * hf] = rmax[r];
    }
    __syncthreads();
    float gmax[8];
    #pragma unroll
    for (int r = 0; r < 8; ++r) {
      float m = red_max[0][r + 8 * hf];
      #pragma unroll
      for (int ww = 1; ww < 8; ++ww) m = fmaxf(m, red_max[ww][r + 8 * hf]);
      gmax[r] = (m < -1e37f) ? 0.0f : m;   // guard all -inf rows
    }

    float rsum[8];
    #pragma unroll
    for (int r = 0; r < 8; ++r) rsum[r] = 0.0f;
    #pragma unroll
    for (int tt = 0; tt < 8; ++tt) {
      #pragma unroll
      for (int r = 0; r < 8; ++r) {
        float e = __expf(logits[tt][r] - gmax[r]);
        logits[tt][r] = e;
        rsum[r] += e;
      }
    }
    #pragma unroll
    for (int r = 0; r < 8; ++r) {
      float s = rsum[r];
      #pragma unroll
      for (int off = 1; off < 16; off <<= 1) s += __shfl_xor(s, off, 32);
      rsum[r] = s;
    }
    if (lrow == 0) {
      #pragma unroll
      for (int r = 0; r < 8; ++r) red_sum[w][r + 8 * hf] = rsum[r];
    }
    __syncthreads();
    float ginv[8];
    #pragma unroll
    for (int r = 0; r < 8; ++r) {
      float s = 0.0f;
      #pragma unroll
      for (int ww = 0; ww < 8; ++ww) s += red_sum[ww][r + 8 * hf];
      ginv[r] = (s > 0.0f) ? __builtin_amdgcn_rcpf(s) : 0.0f;
    }

    // normalize: write att (fp32), stash p (f16) in LDS, accumulate entropy
    float rent[8];
    #pragma unroll
    for (int r = 0; r < 8; ++r) rent[r] = 0.0f;
    #pragma unroll
    for (int tt = 0; tt < 8; ++tt) {
      const int j = 16 * (w + 8 * tt) + lrow;
      #pragma unroll
      for (int r = 0; r < 8; ++r) {
        const int i = i0 + r + 8 * hf;
        const float p = logits[tt][r] * ginv[r];
        att[(size_t)((b * H_ + h) * L_ + i) * S_ + j] = p;
        rent[r] += p * __logf(fmaxf(p, EPSV));
        lds_p[(r + 8 * hf) * PS + j] = (half_t)p;
      }
    }
    #pragma unroll
    for (int r = 0; r < 8; ++r) {
      float s = rent[r];
      #pragma unroll
      for (int off = 1; off < 16; off <<= 1) s += __shfl_xor(s, off, 32);
      rent[r] = s;
    }
    if (lrow == 0) {
      #pragma unroll
      for (int r = 0; r < 8; ++r) red_ent[w][r + 8 * hf] = rent[r];
    }
    __syncthreads();
    if (tid < 16) {
      float s = 0.0f;
      #pragma unroll
      for (int ww = 0; ww < 8; ++ww) s += red_ent[ww][tid];
      s = -s;
      if (!(s == s) || fabsf(s) > 1e37f) s = 0.0f;   // nan_to_num
      ent[(size_t)(b * H_ + h) * L_ + i0 + tid] = s;
    }

    // ---- Phase 3: O = P V (f16 WMMA, fp32 accum) ----
    const int nblk  = w & 3;            // e-block 0..3
    const int khalf = w >> 2;           // K half 0..1
    const int e0    = nblk * 16;
    v8f acc2 = {};
    if (use_vt) {
      // B fragment = 16 contiguous halves of vt (B,H,E,S): one 32B read
      const half_t* vb0 = vt + (size_t)((b * H_ + h) * E_ + e0 + lrow) * S_ + 16 * hf;
      for (int k0 = khalf * 512; k0 < khalf * 512 + 512; k0 += 32) {
        const int ab = lrow * PS + k0 + 16 * hf;
        v8h c0 = *(const v8h*)&lds_p[ab];
        v8h c1 = *(const v8h*)&lds_p[ab + 8];
        v16h a;
        #pragma unroll
        for (int t = 0; t < 8; ++t) { a[t] = c0[t]; a[t + 8] = c1[t]; }
        v16h bf = *(const v16h*)(vb0 + k0);
        acc2 = __builtin_amdgcn_wmma_f32_16x16x32_f16(false, a, false, bf,
                                                      (short)0, acc2, false, false);
      }
    } else {
      for (int k0 = khalf * 512; k0 < khalf * 512 + 512; k0 += 32) {
        const int ab = lrow * PS + k0 + 8 * hf;
        v8h c0 = *(const v8h*)&lds_p[ab];
        v8h c1 = *(const v8h*)&lds_p[ab + 16];
        v16h a;
        #pragma unroll
        for (int t = 0; t < 8; ++t) { a[t] = c0[t]; a[t + 8] = c1[t]; }
        v16h bf;
        #pragma unroll
        for (int t = 0; t < 8; ++t) {
          const int row = k0 + 8 * hf + 2 * t + 8 * (t >> 2);
          const float* vb = val + (size_t)((b * S_ + row) * H_ + h) * E_ + e0 + lrow;
          bf[2 * t]     = (half_t)vb[0];
          bf[2 * t + 1] = (half_t)vb[H_ * E_];
        }
        acc2 = __builtin_amdgcn_wmma_f32_16x16x32_f16(false, a, false, bf,
                                                      (short)0, acc2, false, false);
      }
    }
    if (khalf == 1) {
      #pragma unroll
      for (int r = 0; r < 8; ++r) lds_red[nblk][lane][r] = acc2[r];
    }
    __syncthreads();
    if (khalf == 0) {
      #pragma unroll
      for (int r = 0; r < 8; ++r) {
        const float t = acc2[r] + lds_red[nblk][lane][r];
        const int i = i0 + r + 8 * hf;
        Vout[(size_t)((b * L_ + i) * H_ + h) * E_ + e0 + lrow] = t;
      }
    }
    __syncthreads();   // protect lds_p / lds_red before next batch overwrites
  }
}

extern "C" void kernel_launch(void* const* d_in, const int* in_sizes, int n_in,
                              void* d_out, int out_size, void* d_ws, size_t ws_size,
                              hipStream_t stream) {
  const float* q      = (const float*)d_in[0];
  const float* key    = (const float*)d_in[1];
  const float* val    = (const float*)d_in[2];
  const float* pos    = (const float*)d_in[3];
  const float* phi    = (const float*)d_in[4];
  const float* u      = (const float*)d_in[5];
  const float* logt   = (const float*)d_in[6];
  const float* thresh = (const float*)d_in[7];
  const int*   causal = (const int*)d_in[8];

  float* out  = (float*)d_out;
  float* Vout = out;                                   // (B,L,H,E)
  float* att  = out + (size_t)B_ * L_ * H_ * E_;       // (B,H,L,S)
  float* ent  = att + (size_t)B_ * H_ * L_ * S_;       // (B,H,L)

  const size_t vt_bytes = (size_t)B_ * H_ * E_ * S_ * sizeof(half_t);  // 4 MB
  const int use_vt = (ws_size >= vt_bytes) ? 1 : 0;
  half_t* vt = (half_t*)d_ws;

  if (use_vt) {
    // (B,S,H,E) f32 -> (B,H,E,S) f16: B*H * 32 s-tiles * 2 e-tiles = 2048 WGs
    val_xpose_kernel<<<B_ * H_ * 32 * 2, 256, 0, stream>>>(val, vt);
  }
  const int blocks = H_ * (L_ / 16);                   // 512 WGs, each does all B
  phisoftmax_kernel<<<blocks, 256, 0, stream>>>(q, key, val, pos, phi, u,
                                                logt, thresh, causal,
                                                vt, use_vt,
                                                Vout, att, ent);
}
